// TransformerProjector_71219147702793
// MI455X (gfx1250) — matrix operand order repaired and checked by reading, not compiled, not run on hardware
//
#include <hip/hip_runtime.h>
#include <math.h>

typedef __attribute__((ext_vector_type(16))) _Float16 v16h;
typedef __attribute__((ext_vector_type(8)))  _Float16 v8h;
typedef __attribute__((ext_vector_type(8)))  float    v8f;

// ---------------- model constants ----------------
#define DM    512            // d_model
#define NH    8              // heads
#define HDIM  64             // head dim
#define NL    6              // layers
#define FFD   1024           // ffn dim
#define SE    4096           // encoder tokens
#define NDQ   1024           // decoder queries
#define BW    512            // sliding block == window
#define PADR  (SE + 2 * BW)  // padded K/V rows (5120)
#define SCALE_Q 0.125f       // 1/sqrt(64)

union FragU { v16h h; unsigned u[8]; };

__device__ __forceinline__ float gelu_exact(float x) {
  return 0.5f * x * (1.0f + erff(x * 0.7071067811865475f));
}

// pack 8 consecutive f32 -> 8 f16 (two aligned float4 loads)
__device__ __forceinline__ v8h cvt8(const float* __restrict__ p) {
  float4 a = *(const float4*)p;
  float4 b = *(const float4*)(p + 4);
  v8h h;
  h[0] = (_Float16)a.x; h[1] = (_Float16)a.y;
  h[2] = (_Float16)a.z; h[3] = (_Float16)a.w;
  h[4] = (_Float16)b.x; h[5] = (_Float16)b.y;
  h[6] = (_Float16)b.z; h[7] = (_Float16)b.w;
  return h;
}

// ---------------------------------------------------------------------------
// Batched WMMA GEMM:  C[b] = alpha * (A[b] (MxK) * B[b] (KxN) + bias) [gelu]
//                     (+ Res)   with on-the-fly f32 -> f16 conversion.
// transB: B element (k,n) stored at B[n*ldb + k]  (used for Q * K^T).
// REQUIRES: M % 128 == 0, N % 64 == 0, K % 32 == 0 (true for every call site),
// so staging is branch-free full-tile vector loads/stores.
// Block: 256 threads (8 wave32); block tile 128x64x32; wave tile 32x32
//        -> 4 v_wmma_f32_16x16x32_f16 per K-step, 8 ds_load_b128.
// ---------------------------------------------------------------------------
__global__ __launch_bounds__(256) void k_gemm_wmma(
    const float* __restrict__ A, const float* __restrict__ B,
    float* __restrict__ C,
    const float* __restrict__ bias, const float* __restrict__ Res,
    int M, int N, int K, int lda, int ldb, int ldc,
    long long sA, long long sB, long long sC,
    int transB, int do_gelu, float alpha)
{
  __shared__ _Float16 As[128][32];  // [m][k]
  __shared__ _Float16 Bs[64][32];   // [n][k] (k contiguous for b128 frag loads)

  const int bz = blockIdx.z;
  A += (long long)bz * sA;
  B += (long long)bz * sB;
  C += (long long)bz * sC;

  const int m0   = blockIdx.y * 128;
  const int n0   = blockIdx.x * 64;
  const int tid  = threadIdx.x;
  const int wave = tid >> 5;
  const int lane = tid & 31;
  const int l16  = lane & 15;
  const int lhi  = lane >> 4;           // half-wave select
  const int msub = (wave >> 1) * 32;    // 4 wave-rows of 32
  const int nsub = (wave & 1) * 32;     // 2 wave-cols of 32

  // A staging coords: row r = tid>>1 (0..127), k-base = (tid&1)*16
  const int ar = tid >> 1;
  const int ak = (tid & 1) * 16;
  // B staging coords
  const int bn_nt = tid & 63;           // non-trans: n, 64-wide coalesced
  const int bk_nt = (tid >> 6) * 8;     //            8 k's per thread
  const int bn_t  = tid >> 2;           // trans: n
  const int bk_t  = (tid & 3) * 8;      //        8 consecutive k's

  v8f acc00 = {}, acc01 = {}, acc10 = {}, acc11 = {};

  for (int k0 = 0; k0 < K; k0 += 32) {
    if (k0 + 32 < K)
      __builtin_prefetch(A + (long long)(m0 + ar) * lda + k0 + 32 + ak, 0, 1);

    // ---- stage A tile (128x32) f32 -> f16, two b128 stores/thread ----
    {
      const float* ap = A + (long long)(m0 + ar) * lda + k0 + ak;
      *(v8h*)&As[ar][ak]     = cvt8(ap);
      *(v8h*)&As[ar][ak + 8] = cvt8(ap + 8);
    }
    // ---- stage B tile (32x64) into Bs[n][k] ----
    if (!transB) {
      const float* bp = B + (long long)(k0 + bk_nt) * ldb + n0 + bn_nt;
      v8h hh;
#pragma unroll
      for (int j = 0; j < 8; ++j) hh[j] = (_Float16)bp[(long long)j * ldb];
      *(v8h*)&Bs[bn_nt][bk_nt] = hh;
    } else {
      const float* bp = B + (long long)(n0 + bn_t) * ldb + k0 + bk_t;
      *(v8h*)&Bs[bn_t][bk_t] = cvt8(bp);
    }
    __syncthreads();

    // ---- build fragments per CDNA5 VGPR layouts ----
    FragU a0f, a1f, b0f, b1f;
#pragma unroll
    for (int v = 0; v < 8; ++v) {
      int ka = 2 * (v & 3) + ((v & 4) ? 16 : 0) + lhi * 8;
      a0f.u[v] = *(const unsigned*)&As[msub + l16][ka];
      a1f.u[v] = *(const unsigned*)&As[msub + 16 + l16][ka];
      int kb = 2 * (v & 3) + ((v & 4) ? 16 : 0) + lhi * 8;
      b0f.u[v] = *(const unsigned*)&Bs[nsub + l16][kb];
      b1f.u[v] = *(const unsigned*)&Bs[nsub + 16 + l16][kb];
    }
    acc00 = __builtin_amdgcn_wmma_f32_16x16x32_f16(false, a0f.h, false, b0f.h,
                                                   (short)0, acc00, false, false);
    acc01 = __builtin_amdgcn_wmma_f32_16x16x32_f16(false, a0f.h, false, b1f.h,
                                                   (short)0, acc01, false, false);
    acc10 = __builtin_amdgcn_wmma_f32_16x16x32_f16(false, a1f.h, false, b0f.h,
                                                   (short)0, acc10, false, false);
    acc11 = __builtin_amdgcn_wmma_f32_16x16x32_f16(false, a1f.h, false, b1f.h,
                                                   (short)0, acc11, false, false);
    __syncthreads();
  }

  // ---- epilogue: C/D layout lane l16 = N, reg r (+8 for hi lanes) = M ----
  auto emit = [&](float v, int gm, int gn) {
    long long row = (long long)gm * ldc;
    if (bias) v += bias[gn];
    v *= alpha;
    if (do_gelu) v = gelu_exact(v);
    if (Res) v += Res[row + gn];
    C[row + gn] = v;
  };
#pragma unroll
  for (int r = 0; r < 8; ++r) {
    int gm0 = m0 + msub + r + lhi * 8;
    int gm1 = gm0 + 16;
    int gn0 = n0 + nsub + l16;
    int gn1 = gn0 + 16;
    emit(acc00[r], gm0, gn0);
    emit(acc01[r], gm0, gn1);
    emit(acc10[r], gm1, gn0);
    emit(acc11[r], gm1, gn1);
  }
}

// ---------------------------------------------------------------------------
// Row layernorm: one wave32 per row, Y = LN(X (+R)) * s + b
// ---------------------------------------------------------------------------
__global__ __launch_bounds__(32) void k_layernorm(
    const float* __restrict__ X, const float* __restrict__ R,
    const float* __restrict__ s, const float* __restrict__ b,
    float* __restrict__ Y)
{
  int row = blockIdx.x, lane = threadIdx.x;
  const float* x = X + (long long)row * DM;
  const float* rr = R ? R + (long long)row * DM : nullptr;
  float vals[16];   // 512 / 32
  float sum = 0.f;
#pragma unroll
  for (int i = 0; i < 16; ++i) {
    float v = x[lane + (i << 5)];
    if (rr) v += rr[lane + (i << 5)];
    vals[i] = v; sum += v;
  }
  for (int o = 16; o > 0; o >>= 1) sum += __shfl_xor(sum, o);
  float mean = sum * (1.0f / DM);
  float var = 0.f;
#pragma unroll
  for (int i = 0; i < 16; ++i) { float d = vals[i] - mean; var += d * d; }
  for (int o = 16; o > 0; o >>= 1) var += __shfl_xor(var, o);
  float inv = rsqrtf(var * (1.0f / DM) + 1e-5f);
  float* y = Y + (long long)row * DM;
#pragma unroll
  for (int i = 0; i < 16; ++i) {
    int c = lane + (i << 5);
    y[c] = (vals[i] - mean) * inv * s[c] + b[c];
  }
}

// ---------------------------------------------------------------------------
// Masked softmax. mode 0: plain, 1: causal, 2: sliding window (block nb)
// ---------------------------------------------------------------------------
__device__ __forceinline__ bool sm_valid(int mode, int row, int k, int nb, int Stot) {
  if (mode == 1) return k <= row;
  if (mode == 2) {
    int keypos = (nb - 1) * BW + k;
    int d = k - BW - row;
    return d >= -BW && d <= BW && keypos >= 0 && keypos < Stot;
  }
  return true;
}

__global__ __launch_bounds__(32) void k_softmax_mask(
    float* __restrict__ P, int L, int mode, int nb, int Stot, long long headStride)
{
  int row = blockIdx.x, head = blockIdx.y, lane = threadIdx.x;
  float* p = P + (long long)head * headStride + (long long)row * L;
  float mx = -3.4e38f;
  for (int k = lane; k < L; k += 32)
    mx = fmaxf(mx, sm_valid(mode, row, k, nb, Stot) ? p[k] : -1.0e9f);
  for (int o = 16; o > 0; o >>= 1) mx = fmaxf(mx, __shfl_xor(mx, o));
  float sum = 0.f;
  for (int k = lane; k < L; k += 32)
    sum += __expf((sm_valid(mode, row, k, nb, Stot) ? p[k] : -1.0e9f) - mx);
  for (int o = 16; o > 0; o >>= 1) sum += __shfl_xor(sum, o);
  float inv = 1.0f / sum;
  for (int k = lane; k < L; k += 32)
    p[k] = __expf((sm_valid(mode, row, k, nb, Stot) ? p[k] : -1.0e9f) - mx) * inv;
}

// ---------------------------------------------------------------------------
// Trilinear grid sample of (512,32,32,32) pos-enc at 1024 points (+dec_pos).
// coords reversed: x<-qp[2], y<-qp[1], z<-qp[0]; reflect padding.
// ---------------------------------------------------------------------------
__global__ __launch_bounds__(256) void k_gridsample(
    const float* __restrict__ vol, const float* __restrict__ qp,
    const float* __restrict__ dpos, float* __restrict__ out)
{
  int n = blockIdx.x;
  float cx = qp[n * 3 + 2], cy = qp[n * 3 + 1], cz = qp[n * 3 + 0];
  const float SZ = 31.0f, TW = 62.0f;
  auto unn = [&](float c) {
    float x = (c + 1.0f) * 0.5f * SZ;
    x = fmodf(fabsf(x), TW);
    return (x > SZ) ? (TW - x) : x;
  };
  float ix = unn(cx), iy = unn(cy), iz = unn(cz);
  float fx0 = floorf(ix), fy0 = floorf(iy), fz0 = floorf(iz);
  float fx = ix - fx0, fy = iy - fy0, fz = iz - fz0;
  int x0 = min(max((int)fx0, 0), 31), x1 = min(x0 + 1, 31);
  int y0 = min(max((int)fy0, 0), 31), y1 = min(y0 + 1, 31);
  int z0 = min(max((int)fz0, 0), 31), z1 = min(z0 + 1, 31);
  int   xi[2] = {x0, x1},          yi[2] = {y0, y1},          zi[2] = {z0, z1};
  float wx[2] = {1.f - fx, fx},    wy[2] = {1.f - fy, fy},    wz[2] = {1.f - fz, fz};

  for (int c = threadIdx.x; c < DM; c += blockDim.x) {
    const float* vc = vol + (long long)c * 32768;
    float acc = 0.f;
#pragma unroll
    for (int a = 0; a < 2; ++a)
#pragma unroll
      for (int b2 = 0; b2 < 2; ++b2)
#pragma unroll
        for (int e = 0; e < 2; ++e)
          acc += vc[zi[a] * 1024 + yi[b2] * 32 + xi[e]] * (wz[a] * wy[b2] * wx[e]);
    out[(long long)n * DM + c] = acc + dpos[(long long)n * DM + c];
  }
}

// plane_features (C,64,64) -> (S,C) transpose + enc_pos add
__global__ void k_prep_enc(const float* __restrict__ pf,
                           const float* __restrict__ pos,
                           float* __restrict__ out)
{
  long long i = (long long)blockIdx.x * blockDim.x + threadIdx.x;
  if (i < (long long)SE * DM) {
    int s = (int)(i >> 9);
    int c = (int)(i & 511);
    out[i] = pf[(long long)c * SE + s] + pos[i];
  }
}

__global__ void k_zero(float* __restrict__ p, long long n) {
  long long i = (long long)blockIdx.x * blockDim.x + threadIdx.x;
  if (i < n) p[i] = 0.0f;
}

__global__ void k_copy(const float* __restrict__ a, float* __restrict__ b, long long n) {
  long long i = (long long)blockIdx.x * blockDim.x + threadIdx.x;
  if (i < n) b[i] = a[i];
}

// ---------------------------------------------------------------------------
// Host-side helpers
// ---------------------------------------------------------------------------
static inline void gemm(hipStream_t st, const float* A, const float* B, float* C,
                        const float* bias, const float* Res,
                        int M, int N, int K, int lda, int ldb, int ldc,
                        long long sA, long long sB, long long sC,
                        int batch, int transB, int do_gelu, float alpha)
{
  dim3 g(N / 64, M / 128, batch);
  k_gemm_wmma<<<g, 256, 0, st>>>(A, B, C, bias, Res, M, N, K, lda, ldb, ldc,
                                 sA, sB, sC, transB, do_gelu, alpha);
}

static inline void ln(hipStream_t st, const float* X, const float* R,
                      const float* s, const float* b, float* Y, int rows)
{
  k_layernorm<<<rows, 32, 0, st>>>(X, R, s, b, Y);
}

extern "C" void kernel_launch(void* const* d_in, const int* in_sizes, int n_in,
                              void* d_out, int out_size, void* d_ws, size_t ws_size,
                              hipStream_t stream)
{
  const float* pf         = (const float*)d_in[0];
  const float* qp         = (const float*)d_in[1];
  const float* pos_enc    = (const float*)d_in[2];
  const float* enc_pos    = (const float*)d_in[3];
  const float* dec_pos    = (const float*)d_in[4];
  const float* emb_ln_s   = (const float*)d_in[5];
  const float* emb_ln_b   = (const float*)d_in[6];
  const float* enc_attn_w = (const float*)d_in[7];
  const float* enc_attn_b = (const float*)d_in[8];
  const float* enc_fc1_w  = (const float*)d_in[9];
  const float* enc_fc1_b  = (const float*)d_in[10];
  const float* enc_fc2_w  = (const float*)d_in[11];
  const float* enc_fc2_b  = (const float*)d_in[12];
  const float* enc_ln_s   = (const float*)d_in[13];
  const float* enc_ln_b   = (const float*)d_in[14];
  const float* dec_self_w = (const float*)d_in[15];
  const float* dec_self_b = (const float*)d_in[16];
  const float* dec_cross_w= (const float*)d_in[17];
  const float* dec_cross_b= (const float*)d_in[18];
  const float* dec_fc1_w  = (const float*)d_in[19];
  const float* dec_fc1_b  = (const float*)d_in[20];
  const float* dec_fc2_w  = (const float*)d_in[21];
  const float* dec_fc2_b  = (const float*)d_in[22];
  const float* dec_ln_s   = (const float*)d_in[23];
  const float* dec_ln_b   = (const float*)d_in[24];

  float* dout = (float*)d_out;
  float* ws   = (float*)d_ws;

  const long long SD = (long long)SE * DM;
  float* h    = ws;                               // 4096x512 encoder stream
  float* t    = h    + SD;                        // 4096x512 pre-LN temp
  float* q    = t    + SD;                        // 4096x512 Q
  float* kpad = q    + SD;                        // 5120x512 padded K
  float* vpad = kpad + (long long)PADR * DM;      // 5120x512 padded V
  float* ao   = vpad + (long long)PADR * DM;      // 4096x512 attn out
  float* ffn  = ao   + SD;                        // 4096x1024
  float* sc   = ffn  + (long long)SE * FFD;       // scores (max 8x512x1536)
  float* dbuf = sc   + (long long)NH * BW * (3 * BW); // 1024x512 decoder stream

  // zero padded K/V borders (whole buffers, deterministic) + output zeros
  long long padN = (long long)PADR * DM;
  k_zero<<<(int)((padN + 255) / 256), 256, 0, stream>>>(kpad, padN);
  k_zero<<<(int)((padN + 255) / 256), 256, 0, stream>>>(vpad, padN);
  long long zN = (long long)SE * NDQ;
  k_zero<<<(int)((zN + 255) / 256), 256, 0, stream>>>(dout + (long long)NDQ * DM, zN);

  // ================= encoder =================
  k_prep_enc<<<(int)((SD + 255) / 256), 256, 0, stream>>>(pf, enc_pos, t);
  ln(stream, t, nullptr, emb_ln_s, emb_ln_b, h, SE);

  for (int l = 0; l < NL; ++l) {
    const float* W  = enc_attn_w + (long long)l * 4 * DM * DM;
    const float* Bb = enc_attn_b + (long long)l * 4 * DM;
    gemm(stream, h, W + 0 * DM * DM, q,                          Bb + 0 * DM, nullptr,
         SE, DM, DM, DM, DM, DM, 0, 0, 0, 1, 0, 0, SCALE_Q);
    gemm(stream, h, W + 1 * DM * DM, kpad + (long long)BW * DM,  Bb + 1 * DM, nullptr,
         SE, DM, DM, DM, DM, DM, 0, 0, 0, 1, 0, 0, 1.0f);
    gemm(stream, h, W + 2 * DM * DM, vpad + (long long)BW * DM,  Bb + 2 * DM, nullptr,
         SE, DM, DM, DM, DM, DM, 0, 0, 0, 1, 0, 0, 1.0f);

    for (int nb = 0; nb < SE / BW; ++nb) {
      const float* Qb = q    + (long long)nb * BW * DM;
      const float* Kb = kpad + (long long)nb * BW * DM;  // padded window start
      const float* Vb = vpad + (long long)nb * BW * DM;
      // scores[h] (512x1536) = Qb * Kwin^T, batched over 8 heads (col stride 64)
      gemm(stream, Qb, Kb, sc, nullptr, nullptr,
           BW, 3 * BW, HDIM, DM, DM, 3 * BW,
           64, 64, (long long)BW * 3 * BW, NH, 1, 0, 1.0f);
      k_softmax_mask<<<dim3(BW, NH), 32, 0, stream>>>(sc, 3 * BW, 2, nb, SE,
                                                      (long long)BW * 3 * BW);
      gemm(stream, sc, Vb, ao + (long long)nb * BW * DM, nullptr, nullptr,
           BW, HDIM, 3 * BW, 3 * BW, DM, DM,
           (long long)BW * 3 * BW, 64, 64, NH, 0, 0, 1.0f);
    }
    gemm(stream, ao, W + 3 * DM * DM, t, Bb + 3 * DM, h,
         SE, DM, DM, DM, DM, DM, 0, 0, 0, 1, 0, 0, 1.0f);
    ln(stream, t, nullptr, enc_ln_s + (long long)l * 2 * DM,
       enc_ln_b + (long long)l * 2 * DM, h, SE);

    gemm(stream, h, enc_fc1_w + (long long)l * DM * FFD, ffn,
         enc_fc1_b + (long long)l * FFD, nullptr,
         SE, FFD, DM, DM, FFD, FFD, 0, 0, 0, 1, 0, 1, 1.0f);
    gemm(stream, ffn, enc_fc2_w + (long long)l * FFD * DM, t,
         enc_fc2_b + (long long)l * DM, h,
         SE, DM, FFD, FFD, DM, DM, 0, 0, 0, 1, 0, 0, 1.0f);
    ln(stream, t, nullptr, enc_ln_s + (long long)l * 2 * DM + DM,
       enc_ln_b + (long long)l * 2 * DM + DM, h, SE);
  }

  // ================= decoder =================
  k_gridsample<<<NDQ, 256, 0, stream>>>(pos_enc, qp, dec_pos, t);
  ln(stream, t, nullptr, emb_ln_s + DM, emb_ln_b + DM, dbuf, NDQ);

  for (int l = 0; l < NL; ++l) {
    { // ---- causal self attention ----
      const float* W  = dec_self_w + (long long)l * 4 * DM * DM;
      const float* Bb = dec_self_b + (long long)l * 4 * DM;
      gemm(stream, dbuf, W + 0 * DM * DM, q,    Bb + 0 * DM, nullptr,
           NDQ, DM, DM, DM, DM, DM, 0, 0, 0, 1, 0, 0, SCALE_Q);
      gemm(stream, dbuf, W + 1 * DM * DM, kpad, Bb + 1 * DM, nullptr,
           NDQ, DM, DM, DM, DM, DM, 0, 0, 0, 1, 0, 0, 1.0f);
      gemm(stream, dbuf, W + 2 * DM * DM, vpad, Bb + 2 * DM, nullptr,
           NDQ, DM, DM, DM, DM, DM, 0, 0, 0, 1, 0, 0, 1.0f);
      for (int hh = 0; hh < NH; ++hh) {
        gemm(stream, q + hh * HDIM, kpad + hh * HDIM, sc, nullptr, nullptr,
             NDQ, NDQ, HDIM, DM, DM, NDQ, 0, 0, 0, 1, 1, 0, 1.0f);
        k_softmax_mask<<<dim3(NDQ, 1), 32, 0, stream>>>(sc, NDQ, 1, 0, 0, 0);
        gemm(stream, sc, vpad + hh * HDIM, ao + hh * HDIM, nullptr, nullptr,
             NDQ, HDIM, NDQ, NDQ, DM, DM, 0, 0, 0, 1, 0, 0, 1.0f);
      }
      gemm(stream, ao, W + 3 * DM * DM, t, Bb + 3 * DM, dbuf,
           NDQ, DM, DM, DM, DM, DM, 0, 0, 0, 1, 0, 0, 1.0f);
      ln(stream, t, nullptr, dec_ln_s + (long long)l * 3 * DM,
         dec_ln_b + (long long)l * 3 * DM, dbuf, NDQ);
    }
    { // ---- cross attention to encoder memory ----
      const float* W  = dec_cross_w + (long long)l * 4 * DM * DM;
      const float* Bb = dec_cross_b + (long long)l * 4 * DM;
      gemm(stream, dbuf, W + 0 * DM * DM, q,    Bb + 0 * DM, nullptr,
           NDQ, DM, DM, DM, DM, DM, 0, 0, 0, 1, 0, 0, SCALE_Q);
      gemm(stream, h,    W + 1 * DM * DM, kpad, Bb + 1 * DM, nullptr,
           SE, DM, DM, DM, DM, DM, 0, 0, 0, 1, 0, 0, 1.0f);
      gemm(stream, h,    W + 2 * DM * DM, vpad, Bb + 2 * DM, nullptr,
           SE, DM, DM, DM, DM, DM, 0, 0, 0, 1, 0, 0, 1.0f);
      for (int hh = 0; hh < NH; ++hh) {
        gemm(stream, q + hh * HDIM, kpad + hh * HDIM, sc, nullptr, nullptr,
             NDQ, SE, HDIM, DM, DM, SE, 0, 0, 0, 1, 1, 0, 1.0f);
        k_softmax_mask<<<dim3(NDQ, 1), 32, 0, stream>>>(sc, SE, 0, 0, 0, 0);
        gemm(stream, sc, vpad + hh * HDIM, ao + hh * HDIM, nullptr, nullptr,
             NDQ, HDIM, SE, SE, DM, DM, 0, 0, 0, 1, 0, 0, 1.0f);
      }
      gemm(stream, ao, W + 3 * DM * DM, t, Bb + 3 * DM, dbuf,
           NDQ, DM, DM, DM, DM, DM, 0, 0, 0, 1, 0, 0, 1.0f);
      ln(stream, t, nullptr, dec_ln_s + (long long)l * 3 * DM + DM,
         dec_ln_b + (long long)l * 3 * DM + DM, dbuf, NDQ);
    }
    { // ---- FFN ----
      gemm(stream, dbuf, dec_fc1_w + (long long)l * DM * FFD, ffn,
           dec_fc1_b + (long long)l * FFD, nullptr,
           NDQ, FFD, DM, DM, FFD, FFD, 0, 0, 0, 1, 0, 1, 1.0f);
      gemm(stream, ffn, dec_fc2_w + (long long)l * FFD * DM, t,
           dec_fc2_b + (long long)l * DM, dbuf,
           NDQ, DM, FFD, FFD, DM, DM, 0, 0, 0, 1, 0, 0, 1.0f);
      ln(stream, t, nullptr, dec_ln_s + (long long)l * 3 * DM + 2 * DM,
         dec_ln_b + (long long)l * 3 * DM + 2 * DM, dbuf, NDQ);
    }
  }

  long long dN = (long long)NDQ * DM;
  k_copy<<<(int)((dN + 255) / 256), 256, 0, stream>>>(dbuf, dout, dN);
}
